// LlamaAttention_13993003451046
// MI455X (gfx1250) — matrix operand order repaired and checked by reading, not compiled, not run on hardware
//
#include <hip/hip_runtime.h>
#include <hip/hip_bf16.h>
#include <math.h>
#include <stdint.h>

typedef __attribute__((ext_vector_type(16))) _Float16 v16h;
typedef __attribute__((ext_vector_type(8)))  _Float16 h8;
typedef __attribute__((ext_vector_type(8)))  float    v8f;
typedef __attribute__((ext_vector_type(4)))  unsigned int u32x4;
typedef __attribute__((ext_vector_type(8)))  int          i32x8;
typedef __attribute__((ext_vector_type(4)))  int          i32x4;

#define HIDDEN 4096
#define NHEADS 32
#define NKV    8
#define HD     128
#define QSZ    4096
#define KVSZ   1024
#define SEQ    2048
#define QKV_N  6144

#if __has_builtin(__builtin_amdgcn_tensor_load_to_lds) && \
    __has_builtin(__builtin_amdgcn_s_wait_tensorcnt)
#define HAVE_TDM 1
#else
#define HAVE_TDM 0
#endif

// ---------------- fragment helpers ----------------
// A 16x32 f16 fragment: lane<16 holds K 0..15, lane>=16 holds K 16..31 (contiguous).
// Caller passes p = row_base + (hi ? 16 : 0); reads 16 contiguous halves.
__device__ __forceinline__ v16h fragA(const _Float16* p) {
  h8 lo = *(const h8*)p;
  h8 hi = *(const h8*)(p + 8);
  v16h r;
#pragma unroll
  for (int e = 0; e < 8; ++e) { r[e] = lo[e]; r[e + 8] = hi[e]; }
  return r;
}
// B 32x16 f16 fragment: lane<16 holds K 0..15, lane>=16 holds K 16..31 (contiguous).
// Caller passes p = col_base + (hi ? 16 : 0); reads 16 contiguous halves.
__device__ __forceinline__ v16h fragB(const _Float16* p) {
  h8 lo = *(const h8*)p;
  h8 hi = *(const h8*)(p + 8);
  v16h r;
#pragma unroll
  for (int e = 0; e < 8; ++e) { r[e] = lo[e]; r[e + 8] = hi[e]; }
  return r;
}

__device__ __forceinline__ v8f v8f_zero() {
  v8f z = {0.f, 0.f, 0.f, 0.f, 0.f, 0.f, 0.f, 0.f};
  return z;
}

#if HAVE_TDM
// Issue a 2D TDM load: tile_d1 rows x tile_d0 f16 elements, row stride
// `stride0` elements in global memory, into LDS at byte offset lds_addr.
// pad_iv: pad after 2^(pad_iv+1) DWORDs; pad_amt: insert (pad_amt+1) DWORDs.
// D# bitfields per CDNA5 ISA ch.8 (group0 128b, group1 256b; groups 2/3 zero).
__device__ __forceinline__ void tdm_load_2d_f16(unsigned lds_addr, const _Float16* gptr,
                                                unsigned tile_d0, unsigned tile_d1,
                                                unsigned long long stride0,
                                                int pad_iv, int pad_amt) {
  unsigned long long ga = (unsigned long long)(size_t)gptr;
  u32x4 g0;
  g0[0] = 1u;                                             // count=1, user descriptor
  g0[1] = lds_addr;                                       // LDS byte address
  g0[2] = (unsigned)(ga & 0xffffffffull);                 // global_addr[31:0]
  g0[3] = (unsigned)((ga >> 32) & 0x1ffffffull) | (2u << 30);  // [56:32] | type=2
  const unsigned long long td0 = 1u << 30;                // huge tensor dims: no OOB clip
  const unsigned long long td1 = 1u << 30;
  i32x8 g1;
  g1[0] = (1 << 16)            // data_size = 1 -> 2 bytes
        | (1 << 20)            // pad_enable
        | (pad_iv << 22)
        | (pad_amt << 25);
  g1[1] = (int)((td0 & 0xffffu) << 16);                            // tensor_dim0[15:0]
  g1[2] = (int)((td0 >> 16) & 0xffffu) | (int)((td1 & 0xffffu) << 16);
  g1[3] = (int)((td1 >> 16) & 0xffffu) | (int)(tile_d0 << 16);     // tile_dim0
  g1[4] = (int)(tile_d1 & 0xffffu);                                // tile_dim1 (tile_dim2=0)
  g1[5] = (int)(stride0 & 0xffffffffull);                          // tensor_dim0_stride lo
  g1[6] = (int)((stride0 >> 32) & 0xffffull);                      // stride hi (dim1_stride=0)
  g1[7] = 0;
  i32x4 z4 = {0, 0, 0, 0};
  i32x8 z8 = {0, 0, 0, 0, 0, 0, 0, 0};
  __builtin_amdgcn_tensor_load_to_lds(g0, g1, z4, z4, z8, 0);
}
#endif

// ---------------- fp32 -> f16 cast ----------------
__global__ void cast_f32_to_f16(const float* __restrict__ in, _Float16* __restrict__ out, int n) {
  int i = blockIdx.x * blockDim.x + threadIdx.x;
  int stride = gridDim.x * blockDim.x;
  for (; i < n; i += stride) out[i] = (_Float16)in[i];
}

// ---------------- tiled transpose + cast: out[c][r] = in[r*ldi + c] ----------
// grid = (C/32, R/32), block = 256 (32x8)
__global__ __launch_bounds__(256) void transpose_cast_f32_to_f16(
    const float* __restrict__ in, _Float16* __restrict__ out,
    int ldi, int ldo) {
  __shared__ float t[32][33];
  const int r0 = blockIdx.y * 32, c0 = blockIdx.x * 32;
  const int tx = threadIdx.x & 31, ty = threadIdx.x >> 5;  // ty 0..7
#pragma unroll
  for (int i = 0; i < 32; i += 8)
    t[ty + i][tx] = in[(size_t)(r0 + ty + i) * ldi + c0 + tx];
  __syncthreads();
#pragma unroll
  for (int i = 0; i < 32; i += 8)
    out[(size_t)(c0 + ty + i) * ldo + r0 + tx] = (_Float16)t[tx][ty + i];
}

// ---------------- WMMA GEMM: C[M,N] = A[M,K] * Bt[N,K]^T, f16 in, f32 out ----
// Both operands row-major with contiguous K -> TDM-stageable 2D tiles.
// Block tile 128x128, K-step 32, 8 waves; wave computes a 32x64 subtile.
// Double-buffered LDS: TDM prefetch of tile k+1 overlaps compute of tile k.
#define BK  32
#define LDT 40   // padded LDS row stride (halves): 80B, conflict-free frag reads

template <int N, int K>
__global__ __launch_bounds__(256) void gemm_f16_wmma(const _Float16* __restrict__ A,
                                                     const _Float16* __restrict__ Bt,
                                                     float* __restrict__ C) {
  __shared__ _Float16 aS[2][128][LDT];  // [buf][m][k]
  __shared__ _Float16 bS[2][128][LDT];  // [buf][n][k]
  const int tid  = threadIdx.x;
  const int lane = tid & 31;
  const int wv   = tid >> 5;       // 0..7
  const int wm   = wv >> 1;        // 0..3 : 32-row strip
  const int wn   = wv & 1;         // 0..1 : 64-col strip
  const int l16  = lane & 15;
  const int hi   = (lane >> 4) & 1;
  const int m0   = blockIdx.y * 128;
  const int n0   = blockIdx.x * 128;

  v8f acc[2][4];
#pragma unroll
  for (int i = 0; i < 2; ++i)
#pragma unroll
    for (int j = 0; j < 4; ++j) acc[i][j] = v8f_zero();

  const _Float16* Ab = A  + (size_t)m0 * K;
  const _Float16* Bb = Bt + (size_t)n0 * K;

#if HAVE_TDM
  const unsigned lds_a0 = (unsigned)(size_t)&aS[0][0][0];
  const unsigned lds_a1 = (unsigned)(size_t)&aS[1][0][0];
  const unsigned lds_b0 = (unsigned)(size_t)&bS[0][0][0];
  const unsigned lds_b1 = (unsigned)(size_t)&bS[1][0][0];
  // prologue: stage tile 0 into buffer 0
  if (wv == 0) {
    tdm_load_2d_f16(lds_a0, Ab, 32, 128, (unsigned long long)K, 3, 3);
    tdm_load_2d_f16(lds_b0, Bb, 32, 128, (unsigned long long)K, 3, 3);
  }
#endif

  int cur = 0;
  for (int k0 = 0; k0 < K; k0 += BK, cur ^= 1) {
    __syncthreads();  // all waves done with the buffer the prefetch will overwrite
#if HAVE_TDM
    if (wv == 0) {
      if (k0 + BK < K) {
        // prefetch next tile into the other buffer, then wait for current tile:
        // TDM ops retire in order -> <=2 outstanding means current pair done.
        tdm_load_2d_f16(cur ? lds_a0 : lds_a1, Ab + k0 + BK, 32, 128,
                        (unsigned long long)K, 3, 3);
        tdm_load_2d_f16(cur ? lds_b0 : lds_b1, Bb + k0 + BK, 32, 128,
                        (unsigned long long)K, 3, 3);
        __builtin_amdgcn_s_wait_tensorcnt(2);
      } else {
        __builtin_amdgcn_s_wait_tensorcnt(0);
      }
    }
#else
#pragma unroll
    for (int it = 0; it < 2; ++it) {
      int idx = tid + it * 256;
      int row = idx >> 2;
      int col = (idx & 3) << 3;
      *(h8*)&aS[cur][row][col] = *(const h8*)(Ab + (size_t)row * K + k0 + col);
    }
#pragma unroll
    for (int it = 0; it < 2; ++it) {
      int idx = tid + it * 256;
      int row = idx >> 2;
      int col = (idx & 3) << 3;
      *(h8*)&bS[cur][row][col] = *(const h8*)(Bb + (size_t)row * K + k0 + col);
    }
#endif
    __syncthreads();  // current buffer ready for all waves

    v16h af[2], bf[4];
#pragma unroll
    for (int i = 0; i < 2; ++i)
      af[i] = fragA(&aS[cur][wm * 32 + i * 16 + l16][hi * 16]);
#pragma unroll
    for (int j = 0; j < 4; ++j)
      bf[j] = fragB(&bS[cur][wn * 64 + j * 16 + l16][hi * 16]);
#pragma unroll
    for (int i = 0; i < 2; ++i)
#pragma unroll
      for (int j = 0; j < 4; ++j)
        acc[i][j] = __builtin_amdgcn_wmma_f32_16x16x32_f16(
            false, af[i], false, bf[j], (short)0, acc[i][j], false, false);
  }

  // C/D layout: lane<16 -> (M=r, N=l16); lane>=16 -> (M=8+r, N=l16)
#pragma unroll
  for (int i = 0; i < 2; ++i)
#pragma unroll
    for (int j = 0; j < 4; ++j) {
      float* cp = C + (size_t)(m0 + wm * 32 + i * 16 + hi * 8) * N
                    + n0 + wn * 64 + j * 16 + l16;
#pragma unroll
      for (int r = 0; r < 8; ++r) cp[(size_t)r * N] = acc[i][j][r];
    }
}

// ---------------- RoPE + split + cast (Q, K only; V handled by transpose) ----
__global__ void rope_split(const float* __restrict__ qkv, const int* __restrict__ pos,
                           _Float16* __restrict__ q16, _Float16* __restrict__ k16) {
  const int s = blockIdx.x;
  const int tid = threadIdx.x;
  const float p = (float)pos[s];
  const float* row = qkv + (size_t)s * QKV_N;
  const float kLn = 9.210340371976184f / 64.0f;  // ln(10000)/64
  const float scale = 0.08838834764831845f;      // 1/sqrt(128), folded into Q
#pragma unroll
  for (int it = 0; it < 8; ++it) {               // Q: 32 heads * 64 pairs
    int pi = tid + it * 256;
    int hh = pi >> 6, i = pi & 63;
    float ang = p * __expf(-kLn * (float)i);
    float c = __cosf(ang), sn = __sinf(ang);
    float x1 = row[hh * HD + i], x2 = row[hh * HD + 64 + i];
    q16[(size_t)s * QSZ + hh * HD + i]      = (_Float16)((x1 * c - x2 * sn) * scale);
    q16[(size_t)s * QSZ + hh * HD + 64 + i] = (_Float16)((x2 * c + x1 * sn) * scale);
  }
#pragma unroll
  for (int it = 0; it < 2; ++it) {               // K: 8 heads * 64 pairs
    int pi = tid + it * 256;
    int hh = pi >> 6, i = pi & 63;
    float ang = p * __expf(-kLn * (float)i);
    float c = __cosf(ang), sn = __sinf(ang);
    const float* kr = row + QSZ;
    float x1 = kr[hh * HD + i], x2 = kr[hh * HD + 64 + i];
    k16[(size_t)s * KVSZ + hh * HD + i]      = (_Float16)(x1 * c - x2 * sn);
    k16[(size_t)s * KVSZ + hh * HD + 64 + i] = (_Float16)(x2 * c + x1 * sn);
  }
}

// ---------------- Flash attention (WMMA, causal, GQA) ----------------
// grid = (SEQ/64 q-tiles, 32 heads); block = 128 thr = 4 waves; 16 q-rows/wave.
// K tile row-major [key][feat]; V pre-transposed globally: vT[c][s], c=kv feat.
#define LK 136   // ks row stride (halves): 256B + 16B pad
#define LV 72    // vt row stride (halves): 128B + 16B pad
#define LP 72    // ps row stride

__global__ __launch_bounds__(128) void attn_fwd(const _Float16* __restrict__ q16,
                                                const _Float16* __restrict__ k16,
                                                const _Float16* __restrict__ vT,
                                                _Float16* __restrict__ attn) {
  __shared__ _Float16 ks[64][LK];      // K tile [key][feat]
  __shared__ _Float16 vt[HD][LV];      // V tile [feat][key]
  __shared__ _Float16 ps[4][16][LP];   // per-wave P staging [m][key]

  const int tid  = threadIdx.x;
  const int lane = tid & 31;
  const int wv   = tid >> 5;     // 0..3
  const int l16  = lane & 15;
  const int hi   = (lane >> 4) & 1;
  const int qt   = blockIdx.x;   // 0..31
  const int h    = blockIdx.y;   // 0..31
  const int kvh  = h >> 2;       // GQA group of 4

  v16h qa[4];
  {
    const _Float16* qp = q16 + (size_t)(qt * 64 + wv * 16 + l16) * QSZ + h * HD;
#pragma unroll
    for (int kc = 0; kc < 4; ++kc) qa[kc] = fragA(qp + kc * 32 + hi * 16);
  }

  v8f o[8];
#pragma unroll
  for (int d = 0; d < 8; ++d) o[d] = v8f_zero();
  float mrow[8], lrow[8];
#pragma unroll
  for (int r = 0; r < 8; ++r) { mrow[r] = -3.0e38f; lrow[r] = 0.0f; }

#if HAVE_TDM
  const unsigned lds_k = (unsigned)(size_t)&ks[0][0];
  const unsigned lds_v = (unsigned)(size_t)&vt[0][0];
#endif

  for (int j = 0; j <= qt; ++j) {
    __syncthreads();  // previous tile fully consumed
#if HAVE_TDM
    if (wv == 0) {
      // K: 64 rows x 128 halves, pad 64DW->+4DW => stride 136 halves
      tdm_load_2d_f16(lds_k, k16 + (size_t)(j * 64) * KVSZ + kvh * HD,
                      128, 64, (unsigned long long)KVSZ, 5, 3);
      // V^T: 128 rows x 64 halves, pad 32DW->+4DW => stride 72 halves
      tdm_load_2d_f16(lds_v, vT + (size_t)(kvh * HD) * SEQ + j * 64,
                      64, 128, (unsigned long long)SEQ, 4, 3);
      __builtin_amdgcn_s_wait_tensorcnt(0);
    }
#else
#pragma unroll
    for (int it = 0; it < 8; ++it) {
      int idx = tid + it * 128;
      int row = idx >> 4;
      int col = (idx & 15) << 3;
      *(h8*)&ks[row][col] =
          *(const h8*)(k16 + (size_t)(j * 64 + row) * KVSZ + kvh * HD + col);
    }
#pragma unroll
    for (int it = 0; it < 8; ++it) {
      int idx = tid + it * 128;
      int row = idx >> 3;
      int col = (idx & 7) << 3;
      *(h8*)&vt[row][col] =
          *(const h8*)(vT + (size_t)(kvh * HD + row) * SEQ + j * 64 + col);
    }
#endif
    __syncthreads();

    // S = Q K^T : 16 rows x 64 keys
    v8f sc[4];
#pragma unroll
    for (int nt = 0; nt < 4; ++nt) {
      sc[nt] = v8f_zero();
#pragma unroll
      for (int kc = 0; kc < 4; ++kc) {
        v16h bf = fragB(&ks[nt * 16 + l16][kc * 32 + hi * 16]);
        sc[nt] = __builtin_amdgcn_wmma_f32_16x16x32_f16(
            false, qa[kc], false, bf, (short)0, sc[nt], false, false);
      }
    }

    // causal mask + per-row max
    float tmax[8];
#pragma unroll
    for (int r = 0; r < 8; ++r) tmax[r] = -3.0e38f;
#pragma unroll
    for (int nt = 0; nt < 4; ++nt) {
      int key = j * 64 + nt * 16 + l16;
#pragma unroll
      for (int r = 0; r < 8; ++r) {
        int qrow = qt * 64 + wv * 16 + r + hi * 8;
        float v = sc[nt][r];
        if (key > qrow) v = -3.0e38f;
        sc[nt][r] = v;
        tmax[r] = fmaxf(tmax[r], v);
      }
    }
#pragma unroll
    for (int r = 0; r < 8; ++r)
#pragma unroll
      for (int off = 1; off < 16; off <<= 1)
        tmax[r] = fmaxf(tmax[r], __shfl_xor(tmax[r], off, 32));

    // online softmax update; stage P (f16) for the P@V transpose
#pragma unroll
    for (int r = 0; r < 8; ++r) {
      float mnew = fmaxf(mrow[r], tmax[r]);
      float scl  = __expf(mrow[r] - mnew);
      mrow[r] = mnew;
      float rs = 0.0f;
#pragma unroll
      for (int nt = 0; nt < 4; ++nt) {
        float pv = __expf(sc[nt][r] - mnew);
        sc[nt][r] = pv;
        rs += pv;
      }
#pragma unroll
      for (int off = 1; off < 16; off <<= 1) rs += __shfl_xor(rs, off, 32);
      lrow[r] = lrow[r] * scl + rs;
#pragma unroll
      for (int d = 0; d < 8; ++d) o[d][r] *= scl;
#pragma unroll
      for (int nt = 0; nt < 4; ++nt)
        ps[wv][r + hi * 8][nt * 16 + l16] = (_Float16)sc[nt][r];
    }
    __syncthreads();  // P visible

    // O += P @ V  (K-dim = 64 keys = 2 wmma chunks)
#pragma unroll
    for (int kc = 0; kc < 2; ++kc) {
      v16h pf = fragA(&ps[wv][l16][kc * 32 + hi * 16]);
#pragma unroll
      for (int d = 0; d < 8; ++d) {
        v16h bf = fragB(&vt[d * 16 + l16][kc * 32 + hi * 16]);
        o[d] = __builtin_amdgcn_wmma_f32_16x16x32_f16(
            false, pf, false, bf, (short)0, o[d], false, false);
      }
    }
  }

  // normalize and store f16 attention output [S, 4096]
#pragma unroll
  for (int d = 0; d < 8; ++d)
#pragma unroll
    for (int r = 0; r < 8; ++r) {
      int qrow = qt * 64 + wv * 16 + r + hi * 8;
      int dd = d * 16 + l16;
      attn[(size_t)qrow * QSZ + h * HD + dd] = (_Float16)(o[d][r] / lrow[r]);
    }
}

// ---------------- launch ----------------
extern "C" void kernel_launch(void* const* d_in, const int* in_sizes, int n_in,
                              void* d_out, int out_size, void* d_ws, size_t ws_size,
                              hipStream_t stream) {
  const int*   positions = (const int*)d_in[0];
  const float* hidden    = (const float*)d_in[1];
  const float* w_qkv     = (const float*)d_in[2];
  const float* w_o       = (const float*)d_in[3];
  float* out = (float*)d_out;

  char* ws = (char*)d_ws;
  size_t off = 0;
  auto carve = [&](size_t bytes) -> char* {
    char* p = ws + off;
    off += (bytes + 255) & ~(size_t)255;
    return p;
  };
  _Float16* x16    = (_Float16*)carve((size_t)SEQ * HIDDEN * 2);
  _Float16* wqkvT  = (_Float16*)carve((size_t)HIDDEN * QKV_N * 2);  // [N=6144][K=4096]
  _Float16* woT    = (_Float16*)carve((size_t)QSZ * HIDDEN * 2);    // [N=4096][K=4096]
  _Float16* q16    = (_Float16*)carve((size_t)SEQ * QSZ * 2);
  _Float16* k16    = (_Float16*)carve((size_t)SEQ * KVSZ * 2);
  _Float16* vTb    = (_Float16*)carve((size_t)KVSZ * SEQ * 2);      // [1024][2048]
  _Float16* attn16 = (_Float16*)carve((size_t)SEQ * QSZ * 2);
  float*    qkvf   = (float*)carve((size_t)SEQ * QKV_N * 4);

  cast_f32_to_f16<<<2048, 256, 0, stream>>>(hidden, x16, SEQ * HIDDEN);
  // w_qkv [K=4096][N=6144] -> wqkvT [N][K]
  transpose_cast_f32_to_f16<<<dim3(QKV_N / 32, HIDDEN / 32), 256, 0, stream>>>(
      w_qkv, wqkvT, QKV_N, HIDDEN);
  // w_o [K=4096][N=4096] -> woT [N][K]
  transpose_cast_f32_to_f16<<<dim3(HIDDEN / 32, QSZ / 32), 256, 0, stream>>>(
      w_o, woT, HIDDEN, QSZ);

  // QKV projection: [2048,4096] x [4096,6144] -> f32
  gemm_f16_wmma<QKV_N, HIDDEN><<<dim3(QKV_N / 128, SEQ / 128), 256, 0, stream>>>(
      x16, wqkvT, qkvf);

  rope_split<<<SEQ, 256, 0, stream>>>(qkvf, positions, q16, k16);

  // V slice of qkvf [SEQ][6144] (cols 5120..6143) -> vT [1024][2048] (cast)
  transpose_cast_f32_to_f16<<<dim3(KVSZ / 32, SEQ / 32), 256, 0, stream>>>(
      qkvf + (QSZ + KVSZ), vTb, QKV_N, SEQ);

  attn_fwd<<<dim3(SEQ / 64, NHEADS), 128, 0, stream>>>(q16, k16, vTb, attn16);

  // O projection: [2048,4096] x [4096,4096] -> d_out (f32)
  gemm_f16_wmma<HIDDEN, QSZ><<<dim3(HIDDEN / 128, SEQ / 128), 256, 0, stream>>>(
      attn16, woT, out);
}
